// CandidateRetriever_3985729650866
// MI455X (gfx1250) — matrix operand order repaired and checked by reading, not compiled, not run on hardware
//
#include <hip/hip_runtime.h>
#include <hip/hip_bf16.h>
#include <math.h>

// ---------------- problem constants (match reference) ----------------
#define B_Q     256      // queries
#define DIM     64       // embedding dim
#define TOPK    100      // k (reference k=100; device scalar input ignored)
#define NBINS   2048     // raw-bits>>21: positives 0..511(..1023), negatives 1024..2047
#define POSBINS 1024     // positive-float region scanned for the threshold
#define BIN_SHIFT 21
#define CAP     2048     // per-query candidate capacity
#define TILE_N  128      // table rows staged per block iteration
#define LDS_STRIDE 72    // 64 + 8 pad (bf16 elems) -> conflict-free-ish B reads
#define SORTN   2048

typedef __attribute__((ext_vector_type(16))) __bf16 v16bf;
typedef __attribute__((ext_vector_type(8)))  __bf16 v8bf;
typedef __attribute__((ext_vector_type(8)))  float  v8f;

union BF16Frag { v16bf v; v8bf h[2]; };

// float -> bf16, round-to-nearest-even
__device__ inline unsigned short f2bf(float f) {
    unsigned u = __float_as_uint(f);
    unsigned r = u + 0x7FFFu + ((u >> 16) & 1u);
    return (unsigned short)(r >> 16);
}

__device__ inline uint2 pack4(float4 v, float sc) {
    unsigned a = (unsigned)f2bf(v.x * sc) | ((unsigned)f2bf(v.y * sc) << 16);
    unsigned b = (unsigned)f2bf(v.z * sc) | ((unsigned)f2bf(v.w * sc) << 16);
    return make_uint2(a, b);
}

// A fragment (16x32 bf16): lane<16 -> K {0..7,16..23}, lane>=16 -> K {8..15,24..31}
__device__ inline v16bf load_a_frag(const unsigned short* qn, int qt, int kf, int lane) {
    int m  = lane & 15;
    int k0 = kf * 32 + ((lane < 16) ? 0 : 8);
    const unsigned short* base = qn + (size_t)(qt * 16 + m) * DIM;
    BF16Frag f;
    f.h[0] = *(const v8bf*)(const void*)(base + k0);
    f.h[1] = *(const v8bf*)(const void*)(base + k0 + 16);
    return f.v;
}

// B fragment (32x16 bf16) from LDS-staged rows: lane<16 -> K {0..7,16..23}, lane>=16 -> K {8..15,24..31}
__device__ inline v16bf load_b_frag(const unsigned short* s, int rowbase16, int kf, int lane) {
    int n  = lane & 15;
    int k0 = kf * 32 + ((lane < 16) ? 0 : 8);
    const unsigned short* base = s + (size_t)(rowbase16 + n) * LDS_STRIDE + k0;
    const v8bf* p = (const v8bf*)(const void*)base;
    BF16Frag f;
    f.h[0] = p[0];
    f.h[1] = p[2];
    return f.v;
}

// ---------------- kernel 1: normalize queries -> bf16 ----------------
__global__ __launch_bounds__(256) void norm_q_kernel(const float* __restrict__ q,
                                                     unsigned short* __restrict__ qn) {
    int t = threadIdx.x;                 // one query per thread
    const float4* src = (const float4*)(q + (size_t)t * DIM);
    float4 v[16];
    float ss = 0.f;
#pragma unroll
    for (int j = 0; j < 16; ++j) {
        v[j] = src[j];
        ss += v[j].x * v[j].x + v[j].y * v[j].y + v[j].z * v[j].z + v[j].w * v[j].w;
    }
    float sc = 1.0f / fmaxf(sqrtf(ss), 1e-12f);
    uint2* dst = (uint2*)(qn + (size_t)t * DIM);
#pragma unroll
    for (int j = 0; j < 16; ++j) dst[j] = pack4(v[j], sc);
}

// ---------------- kernels 2 & 4: fused normalize + WMMA scores ----------------
// PHASE 0: branchless per-query histogram of raw score bits (forward tile order).
// PHASE 1: collect candidates with score >= float threshold (REVERSE order -> L2 reuse).
template <int PHASE>
__global__ __launch_bounds__(256) void score_kernel(const float* __restrict__ table,
                                                    const unsigned short* __restrict__ qn,
                                                    unsigned* __restrict__ hist,
                                                    const float* __restrict__ thrf,
                                                    unsigned* __restrict__ cnt,
                                                    float* __restrict__ cs,
                                                    unsigned* __restrict__ ci,
                                                    int nrows) {
    __shared__ unsigned short srows[TILE_N * LDS_STRIDE];   // 18 KB staged bf16 rows

    int tid  = threadIdx.x;
    int w    = tid >> 5;      // wave 0..7 -> owns query tiles 2w, 2w+1
    int lane = tid & 31;
    int qhalf = (lane < 16) ? 0 : 8;

    // Query fragments are invariant across table tiles: preload to registers.
    v16bf a00 = load_a_frag(qn, w * 2 + 0, 0, lane);
    v16bf a01 = load_a_frag(qn, w * 2 + 0, 1, lane);
    v16bf a10 = load_a_frag(qn, w * 2 + 1, 0, lane);
    v16bf a11 = load_a_frag(qn, w * 2 + 1, 1, lane);

    // per-lane 32-bit histogram index bases (q * NBINS); r*NBINS folds into imm offset
    int qbase0 = (w * 2 + 0) * 16 + qhalf;
    int qbase1 = (w * 2 + 1) * 16 + qhalf;
    int hofs0  = qbase0 << 11;            // * NBINS
    int hofs1  = qbase1 << 11;

    // PHASE 1: preload per-lane float threshold edges (16 queries per lane)
    float tf[2][8];
    if (PHASE == 1) {
#pragma unroll
        for (int r = 0; r < 8; ++r) {
            tf[0][r] = thrf[qbase0 + r];
            tf[1][r] = thrf[qbase1 + r];
        }
    }

    int ntile = (nrows + TILE_N - 1) / TILE_N;
    for (int it = blockIdx.x; it < ntile; it += gridDim.x) {
        int tile = (PHASE == 0) ? it : (ntile - 1 - it);   // reverse pass re-hits L2
        int tb = tile * TILE_N;

        // ---- stage 1: load 128 rows, L2-normalize, convert bf16 -> LDS ----
        {
            int rl = tid >> 1;          // local row, 2 threads per row
            int h  = tid & 1;           // which half of the row
            int grow = tb + rl;
            float4 v[8];
            float ss = 0.f;
            if (grow < nrows) {
                const float4* src = (const float4*)(table + (size_t)grow * DIM + h * 32);
#pragma unroll
                for (int j = 0; j < 8; ++j) {
                    v[j] = src[j];
                    ss += v[j].x * v[j].x + v[j].y * v[j].y + v[j].z * v[j].z + v[j].w * v[j].w;
                }
            } else {
#pragma unroll
                for (int j = 0; j < 8; ++j) v[j] = make_float4(0.f, 0.f, 0.f, 0.f);
            }
            float tot = ss + __shfl_xor(ss, 1);
            float sc  = (grow < nrows) ? (1.0f / fmaxf(sqrtf(tot), 1e-12f)) : 0.0f;
            uint2* dst = (uint2*)(srows + (size_t)rl * LDS_STRIDE + h * 32);
#pragma unroll
            for (int j = 0; j < 8; ++j) dst[j] = pack4(v[j], sc);
        }
        __syncthreads();

        // ---- stage 2: WMMA bf16 score tiles + selection work ----
#pragma unroll
        for (int nt = 0; nt < 8; ++nt) {
            v16bf b0 = load_b_frag(srows, nt * 16, 0, lane);
            v16bf b1 = load_b_frag(srows, nt * 16, 1, lane);
            int n    = lane & 15;
            int grow = tb + nt * 16 + n;
            bool inb = grow < nrows;
#pragma unroll
            for (int qq = 0; qq < 2; ++qq) {
                v8f c = {0.f, 0.f, 0.f, 0.f, 0.f, 0.f, 0.f, 0.f};
                c = __builtin_amdgcn_wmma_f32_16x16x32_bf16(
                        false, (qq == 0 ? a00 : a10), false, b0, (short)0, c, false, false);
                c = __builtin_amdgcn_wmma_f32_16x16x32_bf16(
                        false, (qq == 0 ? a01 : a11), false, b1, (short)0, c, false, false);
                if (PHASE == 0) {
                    // Branchless: raw bits>>21 bins positives 0..1023, negatives 1024..2047,
                    // +0.0 (incl. padding rows) -> bin 0.  All bins live; scan ignores trash.
                    int hofs = (qq == 0) ? hofs0 : hofs1;
#pragma unroll
                    for (int r = 0; r < 8; ++r) {
                        unsigned bin = __float_as_uint(c[r]) >> BIN_SHIFT;
                        atomicAdd(&hist[hofs + r * NBINS + (int)bin], 1u);
                    }
                } else {
                    int qbase = (qq == 0) ? qbase0 : qbase1;
                    // common case: no candidate among the 8 scores -> single branch
                    bool hit = false;
#pragma unroll
                    for (int r = 0; r < 8; ++r) hit |= (c[r] >= tf[qq][r]);
                    if (__any((int)hit)) {
#pragma unroll
                        for (int r = 0; r < 8; ++r) {
                            float s = c[r];
                            if (inb && s >= tf[qq][r]) {
                                int q = qbase + r;
                                unsigned pos = atomicAdd(&cnt[q], 1u);
                                if (pos < CAP) {
                                    cs[(size_t)q * CAP + pos] = s;
                                    ci[(size_t)q * CAP + pos] = (unsigned)grow;
                                }
                            }
                        }
                    }
                }
            }
        }
        __syncthreads();
    }
}

// ---------------- kernel 3: per-query threshold (float bin edge) ----------------
__global__ __launch_bounds__(256) void thr_kernel(const unsigned* __restrict__ hist,
                                                  float* __restrict__ thrf) {
    int q = threadIdx.x;
    unsigned c = 0, t = 0;   // t=0 fallback: collect all s >= 0 (degenerate data)
    for (int b = POSBINS - 1; b >= 1; --b) {       // positive-float region only
        c += hist[q * NBINS + b];
        if (c >= TOPK) { t = (unsigned)b; break; }
    }
    thrf[q] = __uint_as_float(t << BIN_SHIFT);     // lower edge of threshold bin
}

// ---------------- kernel 5: bitonic final top-k per query ----------------
__global__ __launch_bounds__(256) void topk_kernel(const float* __restrict__ cs,
                                                   const unsigned* __restrict__ ci,
                                                   const unsigned* __restrict__ cnt,
                                                   float* __restrict__ out_s,
                                                   int* __restrict__ out_i) {
    __shared__ float ss[SORTN];
    __shared__ int   si[SORTN];
    int q = blockIdx.x, tid = threadIdx.x;
    unsigned c = cnt[q];
    if (c > CAP) c = CAP;
    for (int i = tid; i < SORTN; i += 256) {
        if (i < (int)c) { ss[i] = cs[(size_t)q * CAP + i]; si[i] = (int)ci[(size_t)q * CAP + i]; }
        else            { ss[i] = -__builtin_inff();       si[i] = 0x7FFFFFFF; }
    }
    __syncthreads();
    // total order: score desc, index asc  -> deterministic despite atomic append order
    for (int k = 2; k <= SORTN; k <<= 1) {
        for (int j = k >> 1; j > 0; j >>= 1) {
            for (int i = tid; i < SORTN; i += 256) {
                int l = i ^ j;
                if (l > i) {
                    float s1 = ss[i], s2 = ss[l];
                    int   i1 = si[i], i2 = si[l];
                    bool less_l_i = (s2 > s1) || (s2 == s1 && i2 < i1);
                    bool less_i_l = (s1 > s2) || (s1 == s2 && i1 < i2);
                    bool sw = ((i & k) == 0) ? less_l_i : less_i_l;
                    if (sw) { ss[i] = s2; ss[l] = s1; si[i] = i2; si[l] = i1; }
                }
            }
            __syncthreads();
        }
    }
    for (int i = tid; i < TOPK; i += 256) {
        out_s[(size_t)q * TOPK + i] = ss[i];
        out_i[(size_t)q * TOPK + i] = si[i];
    }
}

// ---------------- host launcher ----------------
extern "C" void kernel_launch(void* const* d_in, const int* in_sizes, int n_in,
                              void* d_out, int out_size, void* d_ws, size_t ws_size,
                              hipStream_t stream) {
    (void)n_in; (void)out_size; (void)ws_size;
    const float* qraw  = (const float*)d_in[0];
    const float* table = (const float*)d_in[1];
    int nrows = in_sizes[1] / DIM;

    char* ws = (char*)d_ws;
    unsigned short* qn   = (unsigned short*)ws;                          //  32 KB
    unsigned*       hist = (unsigned*)(ws + 32768);                      //   2 MB
    float*          thrf = (float*)(ws + 32768 + (size_t)B_Q * NBINS * 4);
    unsigned*       cnt  = (unsigned*)(thrf + B_Q);
    float*          cs   = (float*)(cnt + B_Q);                          //   2 MB
    unsigned*       ci   = (unsigned*)(cs + (size_t)B_Q * CAP);          //   2 MB

    hipMemsetAsync(hist, 0, (size_t)B_Q * NBINS * 4, stream);
    hipMemsetAsync(cnt, 0, (size_t)B_Q * 4, stream);

    norm_q_kernel<<<1, 256, 0, stream>>>(qraw, qn);
    score_kernel<0><<<2048, 256, 0, stream>>>(table, qn, hist, thrf, cnt, cs, ci, nrows);
    thr_kernel<<<1, 256, 0, stream>>>(hist, thrf);
    score_kernel<1><<<2048, 256, 0, stream>>>(table, qn, hist, thrf, cnt, cs, ci, nrows);
    topk_kernel<<<B_Q, 256, 0, stream>>>(cs, ci, cnt, (float*)d_out,
                                         (int*)d_out + (size_t)B_Q * TOPK);
}
